// KMeansClusterAggregator_7610682048681
// MI455X (gfx1250) — matrix operand order repaired and checked by reading, not compiled, not run on hardware
//
#include <hip/hip_runtime.h>
#include <stdint.h>

#define DDIM 512
#define KCL  1024
#define TILE_N 32          // nodes per workgroup
#define BLOCK  64          // 2 wave32s
#define F4_PER_ROW (DDIM / 4)   // 128

typedef __attribute__((ext_vector_type(16))) __bf16 v16bf;
typedef __attribute__((ext_vector_type(8)))  float  v8f;
typedef int v4i __attribute__((vector_size(16)));
typedef __attribute__((address_space(1))) v4i* gv4i_ptr;
typedef __attribute__((address_space(3))) v4i* lv4i_ptr;

#if defined(__HIP_DEVICE_COMPILE__)
#if __has_builtin(__builtin_amdgcn_global_load_async_to_lds_b128)
#define HAVE_ASYNC_LDS 1
#endif
#endif

__device__ __forceinline__ unsigned short f2bf(float f) {
    // round-to-nearest-even f32 -> bf16
    unsigned int u = __float_as_uint(f);
    unsigned int r = 0x7FFFu + ((u >> 16) & 1u);
    return (unsigned short)((u + r) >> 16);
}

// ---------------------------------------------------------------- zero output
__global__ void kmca_zero_kernel(float* __restrict__ out, int n) {
    int i = blockIdx.x * blockDim.x + threadIdx.x;
    if (i < n) out[i] = 0.0f;
}

// ------------------------------------------- gather centers, bf16 + sq-norms
__global__ __launch_bounds__(BLOCK) void kmca_prep_kernel(
    const float* __restrict__ nodes, const int* __restrict__ cind,
    unsigned short* __restrict__ wcent, float* __restrict__ wcsq, int nN) {
    __shared__ float sPart[2];
    const int k = blockIdx.x;
    const int t = threadIdx.x;
    int row = cind[k];
    if (row < 0) row = 0;
    if (row >= nN) row = nN - 1;
    const float4* src = (const float4*)(nodes + (size_t)row * DDIM);
    float ss = 0.0f;
    for (int c = t; c < F4_PER_ROW; c += BLOCK) {
        float4 f = src[c];
        ss += f.x * f.x + f.y * f.y + f.z * f.z + f.w * f.w;
        unsigned int lo = (unsigned int)f2bf(f.x) | ((unsigned int)f2bf(f.y) << 16);
        unsigned int hi = (unsigned int)f2bf(f.z) | ((unsigned int)f2bf(f.w) << 16);
        *(uint2*)(wcent + (size_t)k * DDIM + c * 4) = make_uint2(lo, hi);
    }
    // wave32 reduce, then cross-wave via LDS
    for (int o = 16; o > 0; o >>= 1) ss += __shfl_xor(ss, o, 32);
    if ((t & 31) == 0) sPart[t >> 5] = ss;
    __syncthreads();
    if (t == 0) wcsq[k] = sPart[0] + sPart[1];
}

// ------------------------------- main: WMMA distances + argmin + scatter-add
__global__ __launch_bounds__(BLOCK) void kmca_main_kernel(
    const float* __restrict__ nodes,
    const unsigned short* __restrict__ wcent,   // bf16 [K, D]
    const float* __restrict__ wcsq,             // [K]
    float* __restrict__ out,                    // [K, D]
    int nN) {
    __shared__ unsigned short sBF[TILE_N * DDIM];   // 32KB bf16 node tile
    __shared__ float sCSQ[KCL];                     // 4KB
    __shared__ int   sAssign[TILE_N];
#if HAVE_ASYNC_LDS
    __shared__ float sF32[TILE_N * DDIM];           // 64KB f32 staging
#endif
    const int tid  = threadIdx.x;
    const int base = blockIdx.x * TILE_N;

    for (int i = tid; i < KCL; i += BLOCK) sCSQ[i] = wcsq[i];

#if HAVE_ASYNC_LDS
    // CDNA5 async global->LDS copy of the fp32 node tile (ASYNCcnt tracked)
    for (int it = tid; it < TILE_N * F4_PER_ROW; it += BLOCK) {
        int r = it >> 7, c = it & (F4_PER_ROW - 1);
        int g = base + r; if (g >= nN) g = nN - 1;
        __builtin_amdgcn_global_load_async_to_lds_b128(
            (gv4i_ptr)(nodes + (size_t)g * DDIM + c * 4),
            (lv4i_ptr)(&sF32[r * DDIM + c * 4]),
            0, 0);
    }
#if __has_builtin(__builtin_amdgcn_s_wait_asynccnt)
    __builtin_amdgcn_s_wait_asynccnt(0);
#else
    asm volatile("s_wait_asynccnt 0" ::: "memory");
#endif
    __syncthreads();
    // convert staged f32 -> bf16 tile
    for (int it = tid; it < TILE_N * F4_PER_ROW; it += BLOCK) {
        int e = it * 4;
        float4 f = *(const float4*)(&sF32[e]);
        unsigned int lo = (unsigned int)f2bf(f.x) | ((unsigned int)f2bf(f.y) << 16);
        unsigned int hi = (unsigned int)f2bf(f.z) | ((unsigned int)f2bf(f.w) << 16);
        *(uint2*)(&sBF[e]) = make_uint2(lo, hi);
    }
#else
    // fallback: direct global load + convert
    for (int it = tid; it < TILE_N * F4_PER_ROW; it += BLOCK) {
        int r = it >> 7, c = it & (F4_PER_ROW - 1);
        int g = base + r; if (g >= nN) g = nN - 1;
        float4 f = *(const float4*)(nodes + (size_t)g * DDIM + c * 4);
        unsigned int lo = (unsigned int)f2bf(f.x) | ((unsigned int)f2bf(f.y) << 16);
        unsigned int hi = (unsigned int)f2bf(f.z) | ((unsigned int)f2bf(f.w) << 16);
        *(uint2*)(&sBF[r * DDIM + c * 4]) = make_uint2(lo, hi);
    }
#endif
    __syncthreads();

    const int lane = tid & 31;
    const int wv   = tid >> 5;          // wave id: 0..1
    const int n    = lane & 15;         // A row within block / B node column
    const int h    = lane >> 4;         // half-wave select
    const int nodeLocal = wv * 16 + n;

    const char* Ab = (const char*)wcent;
    float best  = 3.4e38f;
    int   bestk = 0;

    for (int k0 = 0; k0 < KCL; k0 += 16) {
        v8f acc = {};
#pragma unroll
        for (int t = 0; t < 16; ++t) {
            const int dk = t * 32;
            // A fragment: row k0+n, K-chunks {8h..8h+7} and {16+8h..23+8h}
            const uint4* ap = (const uint4*)(Ab + (((size_t)(k0 + n) * DDIM + dk) * 2 + h * 16));
            union { uint4 u[2]; v16bf v; } ua;
            ua.u[0] = ap[0];
            ua.u[1] = ap[2];            // +32 bytes
            // B fragment: node column, K-chunks {8h..8h+7} and {16+8h..23+8h}
            const uint4* bp = (const uint4*)((const char*)sBF +
                               ((size_t)nodeLocal * DDIM + dk + h * 8) * 2);
            union { uint4 u[2]; v16bf v; } ub;
            ub.u[0] = bp[0];
            ub.u[1] = bp[2];
            acc = __builtin_amdgcn_wmma_f32_16x16x32_bf16(
                false, ua.v, false, ub.v, (short)0, acc, false, false);
        }
        // fold tile into running argmin: VGPR v holds row m = 8h+v, col = n
#pragma unroll
        for (int v = 0; v < 8; ++v) {
            const int kk = k0 + 8 * h + v;
            const float d2 = sCSQ[kk] - 2.0f * acc[v];
            if (d2 < best) { best = d2; bestk = kk; }
        }
    }
    // merge the two half-wave candidates for each node column
    {
        float ob  = __shfl_xor(best, 16, 32);
        int   obk = __shfl_xor(bestk, 16, 32);
        if (ob < best || (ob == best && obk < bestk)) { best = ob; bestk = obk; }
    }
    if (h == 0) sAssign[nodeLocal] = bestk;
    __syncthreads();

    // scatter-add fp32 node rows into assigned centers
    for (int e = tid; e < TILE_N * DDIM; e += BLOCK) {
        int r = e >> 9, c = e & (DDIM - 1);
        int g = base + r;
        if (g < nN) {
            atomicAdd(out + (size_t)sAssign[r] * DDIM + c,
                      nodes[(size_t)g * DDIM + c]);
        }
    }
}

extern "C" void kernel_launch(void* const* d_in, const int* in_sizes, int n_in,
                              void* d_out, int out_size, void* d_ws, size_t ws_size,
                              hipStream_t stream) {
    const float* nodes = (const float*)d_in[0];
    const int*   cind  = (const int*)d_in[1];
    float* out = (float*)d_out;

    const int N = in_sizes[0] / DDIM;     // 100000
    const int K = in_sizes[1];            // 1024

    unsigned short* wcent = (unsigned short*)d_ws;                       // bf16 [K,D] = 1MB
    float* wcsq = (float*)((char*)d_ws + (size_t)KCL * DDIM * 2);        // [K]    = 4KB

    // 1) zero output (harness poisons d_out)
    kmca_zero_kernel<<<(out_size + 255) / 256, 256, 0, stream>>>(out, out_size);
    // 2) gather centers -> bf16 + squared norms
    kmca_prep_kernel<<<K, BLOCK, 0, stream>>>(nodes, cind, wcent, wcsq, N);
    // 3) WMMA distance/argmin + fused scatter-add
    kmca_main_kernel<<<(N + TILE_N - 1) / TILE_N, BLOCK, 0, stream>>>(
        nodes, wcent, wcsq, out, N);
}
